// CustomCholeskyLSTM_58884001628631
// MI455X (gfx1250) — compile-verified, matrix operand order checked by reading, not run on hardware
//
#include <hip/hip_runtime.h>
#include <hip/hip_bf16.h>
#include <stdint.h>

// ---------------- problem dims ----------------
#define B_   128
#define S_   512
#define D_   256
#define H_   512
#define G4_  2048   // 4*H
#define K0_  768    // D+H   (layer-0 GEMM K)
#define K1_  1024   // 2*H   (layer-1 GEMM K)
#define OUT_ 1275
#define NWG_ 32     // persistent workgroups (one per 16-wide h-column slice)

// LDS row strides (bf16 elems), padded +8 elems (16B) so row stride mod 256B = 16B
// -> lane l of a b128 fragment read hits banks 4l..4l+3 (conflict-free).
#define LW0_STRIDE (K0_ + 8)   // 776 elems = 97 uint4
#define LW1_STRIDE (K1_ + 8)   // 1032 elems = 129 uint4
#define SMEM_BYTES ((64 * LW0_STRIDE + 64 * LW1_STRIDE) * 2)

typedef __attribute__((ext_vector_type(16))) __bf16 bf16x16;
typedef __attribute__((ext_vector_type(8)))  float  f32x8;

union FragU { bf16x16 v; uint4 q[2]; };

// A/B bf16 fragment: lane holds two contiguous 16-byte chunks at +0 and +16 elems.
// (A 16x32 layout: lane = M%16, lanes<16 hold K 0..7/16..23, lanes>=16 hold K 8..15/24..31;
//  B mirrored with lane = N when weights are stored transposed [N][K].)
__device__ inline bf16x16 load_frag_g(const __hip_bfloat16* p) {
  FragU f;
  const uint4* q = reinterpret_cast<const uint4*>(p);
  f.q[0] = q[0];
  f.q[1] = q[2];   // +16 bf16 elements = +32 bytes
  return f.v;
}

__device__ inline bf16x16 load_frag_lds(const __hip_bfloat16* p) {
  FragU f;
  const uint4* q = reinterpret_cast<const uint4*>(p);
  f.q[0] = q[0];
  f.q[1] = q[2];
  return f.v;
}

__device__ inline float expf_fast(float x) {
  return __builtin_amdgcn_exp2f(x * 1.44269504088896340736f);
}
__device__ inline float sigm(float x) {
  return __builtin_amdgcn_rcpf(1.0f + expf_fast(-x));
}
__device__ inline float tanh_fast(float x) {
  return fmaf(-2.0f, __builtin_amdgcn_rcpf(1.0f + expf_fast(2.0f * x)), 1.0f);
}

// Grid-wide barrier: count + generation sense.
__device__ inline void grid_barrier(unsigned* cnt, volatile unsigned* gen, unsigned nwg) {
  __syncthreads();
  if (threadIdx.x == 0) {
    __threadfence();                       // release our h-buffer stores
    unsigned g = *gen;
    if (atomicAdd(cnt, 1u) == nwg - 1u) {
      *cnt = 0u;
      __threadfence();
      atomicAdd((unsigned*)gen, 1u);
    } else {
      while (*gen == g) { __builtin_amdgcn_s_sleep(2); }
    }
    __threadfence();                       // acquire other WGs' stores
  }
  __syncthreads();
}

// ---------------- weight / input conversion ----------------
// Wt[n][k] = bf16(W[k][n])  (transpose so B-fragment loads are contiguous in K)
__global__ void convert_w(const float* __restrict__ W, __hip_bfloat16* __restrict__ Wt,
                          int K, int N) {
  size_t i = (size_t)blockIdx.x * blockDim.x + threadIdx.x;
  if (i >= (size_t)K * N) return;
  size_t n = i / (size_t)K, k = i % (size_t)K;
  Wt[i] = __float2bfloat16(W[k * (size_t)N + n]);
}

// xbf[t][b][d] = bf16(x[b][t][d])  (time-major so step t's A-block is contiguous)
__global__ void convert_x(const float* __restrict__ x, __hip_bfloat16* __restrict__ xbf) {
  size_t i = (size_t)blockIdx.x * blockDim.x + threadIdx.x;
  if (i >= (size_t)S_ * B_ * D_) return;
  size_t d = i % D_;
  size_t b = (i / D_) % B_;
  size_t t = i / ((size_t)D_ * B_);
  xbf[i] = __float2bfloat16(x[b * (size_t)S_ * D_ + t * (size_t)D_ + d]);
}

// ---------------- persistent 2-layer LSTM ----------------
__global__ __launch_bounds__(256, 1) void lstm_persistent(
    const __hip_bfloat16* __restrict__ xbf,   // [S][B][D] bf16
    const __hip_bfloat16* __restrict__ W0t,   // [2048][768] bf16 (transposed)
    const __hip_bfloat16* __restrict__ W1t,   // [2048][1024] bf16 (transposed)
    const float* __restrict__ b0,             // [2048]
    const float* __restrict__ b1,             // [2048]
    __hip_bfloat16* h0buf,                    // [2][B][H] bf16 (double buffered)
    __hip_bfloat16* h1buf,                    // [2][B][H] bf16
    float* __restrict__ h1out,                // [B][H] fp32 (final h1)
    unsigned* bar)                            // {count, generation}
{
  extern __shared__ char smem[];
  __hip_bfloat16* lw0 = (__hip_bfloat16*)smem;                 // [64][LW0_STRIDE]
  __hip_bfloat16* lw1 = lw0 + 64 * LW0_STRIDE;                 // [64][LW1_STRIDE]

  const int j    = blockIdx.x;          // h-column slice [16j, 16j+16)
  const int tid  = threadIdx.x;
  const int wave = tid >> 5;            // 8 waves: row tile 16*wave..+16
  const int lane = tid & 31;
  const int nl   = lane & 15;           // N within tile (B/C/D lane mapping)
  const int half = lane >> 4;           // K-chunk / M-offset select
  const int m    = wave;
  const int col  = j * 16 + nl;         // global h column owned by this lane
  const int arow = 16 * m + nl;         // A-matrix row read by this lane

  // ---- one-time: stage this WG's weight slices into LDS (resident 512 steps) ----
  {
    uint4*       d0 = (uint4*)lw0;
    const uint4* s0 = (const uint4*)W0t;
    for (int idx = tid; idx < 64 * 96; idx += 256) {        // 96 uint4 per 768-elem row
      int rb = idx / 96, off = idx % 96;
      int grow = (rb >> 4) * H_ + j * 16 + (rb & 15);       // gate-major local rows
      d0[rb * 97 + off] = s0[(size_t)grow * 96 + off];
    }
    uint4*       d1 = (uint4*)lw1;
    const uint4* s1 = (const uint4*)W1t;
    for (int idx = tid; idx < 64 * 128; idx += 256) {       // 128 uint4 per 1024-elem row
      int rb = idx / 128, off = idx % 128;
      int grow = (rb >> 4) * H_ + j * 16 + (rb & 15);
      d1[rb * 129 + off] = s1[(size_t)grow * 128 + off];
    }
  }
  __syncthreads();

  float bias0[4], bias1[4];
#pragma unroll
  for (int g = 0; g < 4; ++g) {
    bias0[g] = b0[g * H_ + col];
    bias1[g] = b1[g * H_ + col];
  }

  const f32x8 z8 = {0.f, 0.f, 0.f, 0.f, 0.f, 0.f, 0.f, 0.f};
  f32x8 c0 = z8, c1 = z8;               // cell state lives in registers

  for (int t = 0; t < S_; ++t) {
    const int rd = t & 1, wr = rd ^ 1;
    const __hip_bfloat16* xrow = xbf   + (size_t)t  * (B_ * D_);
    const __hip_bfloat16* h0r  = h0buf + (size_t)rd * (B_ * H_);
    __hip_bfloat16*       h0w  = h0buf + (size_t)wr * (B_ * H_);
    const __hip_bfloat16* h1r  = h1buf + (size_t)rd * (B_ * H_);
    __hip_bfloat16*       h1w  = h1buf + (size_t)wr * (B_ * H_);

    // ---- layer 0: z0 = [x_t | h0] @ W0,  K = 768 (24 k-tiles of 32) ----
    f32x8 acc[4] = { z8, z8, z8, z8 };
#pragma unroll 4
    for (int kt = 0; kt < 24; ++kt) {
      const __hip_bfloat16* ap = (kt < 8)
          ? (xrow + (size_t)arow * D_ + kt * 32 + half * 8)
          : (h0r  + (size_t)arow * H_ + (kt - 8) * 32 + half * 8);
      bf16x16 a = load_frag_g(ap);
#pragma unroll
      for (int g = 0; g < 4; ++g) {
        bf16x16 bm = load_frag_lds(lw0 + (size_t)(g * 16 + nl) * LW0_STRIDE
                                       + kt * 32 + half * 8);
        acc[g] = __builtin_amdgcn_wmma_f32_16x16x32_bf16(
            false, a, false, bm, (short)0, acc[g], false, false);
      }
    }
#pragma unroll
    for (int r = 0; r < 8; ++r) {
      float zi = acc[0][r] + bias0[0];
      float zf = acc[1][r] + bias0[1];
      float zg = acc[2][r] + bias0[2];
      float zo = acc[3][r] + bias0[3];
      float cn = sigm(zf) * c0[r] + sigm(zi) * tanh_fast(zg);
      c0[r] = cn;
      float h  = sigm(zo) * tanh_fast(cn);
      int Mg   = 16 * m + r + 8 * half;           // C/D layout: vgpr r, lanes>=16 -> +8
      h0w[(size_t)Mg * H_ + col] = __float2bfloat16(h);
    }

    grid_barrier(bar, bar + 1, NWG_);   // h0(t) fully published

    // ---- layer 1: z1 = [h0_new | h1] @ W1,  K = 1024 (32 k-tiles) ----
    acc[0] = z8; acc[1] = z8; acc[2] = z8; acc[3] = z8;
#pragma unroll 4
    for (int kt = 0; kt < 32; ++kt) {
      const __hip_bfloat16* ap = (kt < 16)
          ? (h0w + (size_t)arow * H_ + kt * 32 + half * 8)
          : (h1r + (size_t)arow * H_ + (kt - 16) * 32 + half * 8);
      bf16x16 a = load_frag_g(ap);
#pragma unroll
      for (int g = 0; g < 4; ++g) {
        bf16x16 bm = load_frag_lds(lw1 + (size_t)(g * 16 + nl) * LW1_STRIDE
                                       + kt * 32 + half * 8);
        acc[g] = __builtin_amdgcn_wmma_f32_16x16x32_bf16(
            false, a, false, bm, (short)0, acc[g], false, false);
      }
    }
#pragma unroll
    for (int r = 0; r < 8; ++r) {
      float zi = acc[0][r] + bias1[0];
      float zf = acc[1][r] + bias1[1];
      float zg = acc[2][r] + bias1[2];
      float zo = acc[3][r] + bias1[3];
      float cn = sigm(zf) * c1[r] + sigm(zi) * tanh_fast(zg);
      c1[r] = cn;
      float h  = sigm(zo) * tanh_fast(cn);
      int Mg   = 16 * m + r + 8 * half;
      h1w[(size_t)Mg * H_ + col] = __float2bfloat16(h);
      if (t == S_ - 1) h1out[(size_t)Mg * H_ + col] = h;  // full-precision final state
    }
    // No second barrier needed: buffer swap + next step's barrier order all hazards.
  }
}

// ---------------- final projection: out = h1 @ Wp + bp ----------------
__global__ void proj_kernel(const float* __restrict__ h1, const float* __restrict__ Wp,
                            const float* __restrict__ bp, float* __restrict__ out) {
  int o = blockIdx.x * blockDim.x + threadIdx.x;
  int b = blockIdx.y;
  if (o >= OUT_) return;
  float acc = bp[o];
  const float* hr = h1 + (size_t)b * H_;
  for (int k = 0; k < H_; ++k)
    acc = fmaf(hr[k], Wp[(size_t)k * OUT_ + o], acc);
  out[(size_t)b * OUT_ + o] = acc;
}

// ---------------- host launcher ----------------
extern "C" void kernel_launch(void* const* d_in, const int* in_sizes, int n_in,
                              void* d_out, int out_size, void* d_ws, size_t ws_size,
                              hipStream_t stream) {
  const float* x  = (const float*)d_in[0];
  const float* W0 = (const float*)d_in[1];
  const float* b0 = (const float*)d_in[2];
  const float* W1 = (const float*)d_in[3];
  const float* b1 = (const float*)d_in[4];
  const float* Wp = (const float*)d_in[5];
  const float* bp = (const float*)d_in[6];
  float* out = (float*)d_out;

  char* ws = (char*)d_ws;
  size_t off = 0;
  auto alloc = [&](size_t bytes) -> void* {
    off = (off + 255) & ~(size_t)255;
    void* p = ws + off;
    off += bytes;
    return p;
  };

  unsigned*       bar   = (unsigned*)alloc(256);
  __hip_bfloat16* W0t   = (__hip_bfloat16*)alloc((size_t)G4_ * K0_ * 2);
  __hip_bfloat16* W1t   = (__hip_bfloat16*)alloc((size_t)G4_ * K1_ * 2);
  __hip_bfloat16* xbf   = (__hip_bfloat16*)alloc((size_t)S_ * B_ * D_ * 2);
  __hip_bfloat16* h0buf = (__hip_bfloat16*)alloc((size_t)2 * B_ * H_ * 2);
  __hip_bfloat16* h1buf = (__hip_bfloat16*)alloc((size_t)2 * B_ * H_ * 2);
  float*          h1out = (float*)alloc((size_t)B_ * H_ * 4);
  (void)ws_size; (void)in_sizes; (void)n_in; (void)out_size;

  hipMemsetAsync(bar,   0, 256, stream);
  hipMemsetAsync(h0buf, 0, (size_t)2 * B_ * H_ * 2, stream);
  hipMemsetAsync(h1buf, 0, (size_t)2 * B_ * H_ * 2, stream);

  {
    size_t n = (size_t)K0_ * G4_;
    convert_w<<<(unsigned)((n + 255) / 256), 256, 0, stream>>>(W0, W0t, K0_, G4_);
  }
  {
    size_t n = (size_t)K1_ * G4_;
    convert_w<<<(unsigned)((n + 255) / 256), 256, 0, stream>>>(W1, W1t, K1_, G4_);
  }
  {
    size_t n = (size_t)S_ * B_ * D_;
    convert_x<<<(unsigned)((n + 255) / 256), 256, 0, stream>>>(x, xbf);
  }

  lstm_persistent<<<NWG_, 256, SMEM_BYTES, stream>>>(xbf, W0t, W1t, b0, b1,
                                                     h0buf, h1buf, h1out, bar);

  proj_kernel<<<dim3((OUT_ + 255) / 256, B_), 256, 0, stream>>>(h1out, Wp, bp, out);
}
